// GATMessagePassing_20074677141747
// MI455X (gfx1250) — compile-verified, matrix operand order checked by reading, not run on hardware
//
#include <hip/hip_runtime.h>
#include <hip/hip_bf16.h>

typedef _Float16 v16h __attribute__((ext_vector_type(16)));
typedef _Float16 v8h  __attribute__((ext_vector_type(8)));
typedef _Float16 v4h  __attribute__((ext_vector_type(4)));
typedef float    v8f  __attribute__((ext_vector_type(8)));

#define HDIM    64
#define ATT_IN  176
#define ATT_INP 192      // K padded to multiple of 32
#define ATT_W   144
#define ATT_NT  9        // 144 / 16 N-tiles
#define ATT_KC  6        // 192 / 32 K-chunks
#define UPD_NT1 8        // 128 / 16
#define UPD_NT2 4        // 64  / 16
#define UPD_KC  4        // 128 / 32

// LDS layout sizes (bytes)
#define EDGE_LDS_B   (ATT_NT * ATT_KC * 512 * 2)          // 55296: att_w1 fragments
#define EDGE_LDS_A   (4 * 16 * ATT_INP * 2)               // 24576: per-wave A staging
#define EDGE_LDS_BYTES (EDGE_LDS_B + EDGE_LDS_A)          // 79872

#define NODE_LDS_B1  (UPD_NT1 * UPD_KC * 512 * 2)         // 32768
#define NODE_LDS_B2  (UPD_NT2 * UPD_KC * 512 * 2)         // 16384
#define NODE_LDS_IN  (4 * 16 * 128 * 2)                   // 16384
#define NODE_LDS_U   (4 * 16 * 128 * 2)                   // 16384
#define NODE_LDS_BYTES (NODE_LDS_B1 + NODE_LDS_B2 + NODE_LDS_IN + NODE_LDS_U) // 81920

union AF { v16h v; v8h h[2]; };
union CF { v8f  v; float f[8]; };

__device__ __forceinline__ void atomicMaxF(float* addr, float v) {
    // ordered-int trick; correct with init = -inf
    if (v >= 0.0f) atomicMax((int*)addr, __float_as_int(v));
    else           atomicMin((unsigned int*)addr, (unsigned int)__float_as_int(v));
}

__device__ __forceinline__ v4h cvt4(float4 p) {
    v4h h; h[0] = (_Float16)p.x; h[1] = (_Float16)p.y;
    h[2] = (_Float16)p.z; h[3] = (_Float16)p.w; return h;
}

// ---------------------------------------------------------------------------
// Repack a row-major fp32 weight [OUT][IN] into f16 WMMA B-fragments.
// Fragment (t,k) is the 32x16 (KxN) tile; per-lane 16 contiguous halves:
//   n = t*16 + (lane&15)
//   Koff(e,lane) = (e&7) + ((e&8)<<1) + ((lane>>4)<<3)
// ---------------------------------------------------------------------------
__global__ void pack_weights(const float* __restrict__ W, int OUT, int IN,
                             int kchunks, int total, _Float16* __restrict__ dst)
{
    int idx = blockIdx.x * blockDim.x + threadIdx.x;
    if (idx >= total) return;
    int e    = idx & 15;
    int lane = (idx >> 4) & 31;
    int frag = idx >> 9;
    int kc   = frag % kchunks;
    int t    = frag / kchunks;
    int n    = t * 16 + (lane & 15);
    int koff = (e & 7) + ((e & 8) << 1) + ((lane >> 4) << 3);
    int K    = kc * 32 + koff;
    float v  = (n < OUT && K < IN) ? W[(long)n * IN + K] : 0.0f;
    dst[idx] = (_Float16)v;
}

__global__ void init_nodes(float* __restrict__ nmax, float* __restrict__ nsum,
                           float* __restrict__ agg, int N)
{
    long i = (long)blockIdx.x * blockDim.x + threadIdx.x;
    if (i < (long)N * HDIM) agg[i] = 0.0f;
    if (i < N) { nmax[i] = __int_as_float(0xFF800000); nsum[i] = 0.0f; }
}

// ---------------------------------------------------------------------------
// Edge attention MLP: weights resident in LDS, grid-stride over 16-edge tiles,
// layer 2 fused into the layer-1 WMMA tile loop via register partials.
// ---------------------------------------------------------------------------
__global__ __launch_bounds__(128) void edge_att_kernel(
    const float* __restrict__ x, const float* __restrict__ xs,
    const int* __restrict__ ei, const float* __restrict__ efeat,
    const _Float16* __restrict__ w1f, const float* __restrict__ b1,
    const float* __restrict__ w2, const float* __restrict__ b2,
    float* __restrict__ rawbuf, float* __restrict__ nmax, int E)
{
    extern __shared__ char dsmem[];
    _Float16* ldsB = (_Float16*)dsmem;                       // 55296 B
    _Float16* ldsA = (_Float16*)(dsmem + EDGE_LDS_B);        // 24576 B

    const int lane = threadIdx.x & 31;
    const int wid  = threadIdx.x >> 5;
    _Float16* aBase = ldsA + (size_t)wid * 16 * ATT_INP;     // wave-private

    // stage all att_w1 fragments into LDS once per block
    {
        const uint4* s = (const uint4*)w1f;
        uint4*       d = (uint4*)ldsB;
        for (int i = threadIdx.x; i < EDGE_LDS_B / 16; i += blockDim.x) d[i] = s[i];
    }
    __syncthreads();

    const int ncol = lane & 15;
    const int kh   = (lane >> 4) << 3;
    const int m0   = (lane >> 4) << 3;

    // hoist per-lane bias / w2 values (t fully unrolled below keeps these in regs)
    float biasv[ATT_NT], w2v[ATT_NT];
    #pragma unroll
    for (int t = 0; t < ATT_NT; ++t) {
        biasv[t] = b1[t * 16 + ncol];
        w2v[t]   = w2[t * 16 + ncol];
    }
    const float b2v = b2[0];

    const long ntiles = ((long)E + 15) >> 4;
    for (long tile = (long)blockIdx.x * 4 + wid; tile < ntiles; tile += (long)gridDim.x * 4) {
        const long eb = tile << 4;

        // ---- gather 16 rows, f32 -> f16, float4 vector loads ----
        for (int m = 0; m < 16; ++m) {
            long e = eb + m;
            bool valid = (e < E);
            int s = 0, d = 0;
            if (valid) { s = ei[e]; d = ei[(long)E + e]; }
            _Float16* row = aBase + m * ATT_INP;

            float4 p = make_float4(0.f, 0.f, 0.f, 0.f);
            if (valid) {
                p = (lane < 16) ? *(const float4*)(x + (long)s * 64 + lane * 4)
                                : *(const float4*)(x + (long)d * 64 + (lane - 16) * 4);
            }
            *(v4h*)(row + lane * 4) = cvt4(p);               // cols 0..127

            if (lane < 16) {                                  // cols 128..191 (incl. pad)
                float4 q = make_float4(0.f, 0.f, 0.f, 0.f);
                if (valid) {
                    if      (lane < 4)  q = *(const float4*)(xs + (long)s * 16 + lane * 4);
                    else if (lane < 8)  q = *(const float4*)(xs + (long)d * 16 + (lane - 4) * 4);
                    else if (lane < 12) q = *(const float4*)(efeat + e * 16 + (lane - 8) * 4);
                }
                *(v4h*)(row + 128 + lane * 4) = cvt4(q);
            }
        }
        // wave-private LDS region: hardware DS counters order store->load, no barrier

        // ---- A fragments: lane holds row M = lane&15 ----
        AF a[ATT_KC];
        const _Float16* arow = aBase + ncol * ATT_INP;
        #pragma unroll
        for (int k = 0; k < ATT_KC; ++k) {
            a[k].h[0] = *(const v8h*)(arow + k * 32 + kh);
            a[k].h[1] = *(const v8h*)(arow + k * 32 + kh + 16);
        }

        // ---- fused layer 1 (WMMA) + layer 2 partial dot ----
        float part[8];
        #pragma unroll
        for (int r = 0; r < 8; ++r) part[r] = 0.f;

        #pragma unroll
        for (int t = 0; t < ATT_NT; ++t) {
            CF c; c.v = (v8f){};
            #pragma unroll
            for (int k = 0; k < ATT_KC; ++k) {
                const v16h* bp = (const v16h*)(ldsB + ((t * ATT_KC + k) * 512 + lane * 16));
                c.v = __builtin_amdgcn_wmma_f32_16x16x32_f16(
                          false, a[k].v, false, *bp, (short)0, c.v, false, false);
            }
            #pragma unroll
            for (int r = 0; r < 8; ++r) {
                float v = c.f[r] + biasv[t];
                v = v > 0.f ? v : 0.f;                        // relu(h)
                part[r] = fmaf(v, w2v[t], part[r]);
            }
        }

        // butterfly-reduce partials across the 16 lanes of each half-wave
        #pragma unroll
        for (int off = 8; off >= 1; off >>= 1) {
            #pragma unroll
            for (int r = 0; r < 8; ++r)
                part[r] += __shfl_xor(part[r], off, 32);
        }

        if ((lane & 15) == 0) {                               // lane 0 -> rows 0..7, lane 16 -> rows 8..15
            #pragma unroll
            for (int r = 0; r < 8; ++r) {
                long e = eb + m0 + r;
                if (e < E) {
                    float acc = part[r] + b2v;
                    float raw = acc > 0.f ? acc : 0.01f * acc; // leaky_relu
                    rawbuf[e] = raw;
                    atomicMaxF(&nmax[ei[(long)E + e]], raw);
                }
            }
        }
    }
}

__global__ void edge_exp_kernel(const int* __restrict__ ei, float* __restrict__ raww,
                                const float* __restrict__ nmax, float* __restrict__ nsum,
                                int E)
{
    long e = (long)blockIdx.x * blockDim.x + threadIdx.x;
    if (e < E) {
        int d = ei[(long)E + e];
        float w = __expf(raww[e] - nmax[d]);
        raww[e] = w;
        atomicAdd(&nsum[d], w);
    }
}

__global__ void edge_agg_kernel(const int* __restrict__ ei, const float* __restrict__ raww,
                                const float* __restrict__ nsum, const float* __restrict__ x,
                                float* __restrict__ agg, int E)
{
    long t    = (long)blockIdx.x * blockDim.x + threadIdx.x;
    long e    = t >> 5;
    int  lane = (int)(t & 31);
    if (e < E) {
        int  d     = ei[(long)E + e];
        int  s     = ei[e];
        float attn = raww[e] / (nsum[d] + 1e-9f);
        atomicAdd(&agg[(long)d * 64 + lane],      x[(long)s * 64 + lane]      * attn);
        atomicAdd(&agg[(long)d * 64 + lane + 32], x[(long)s * 64 + lane + 32] * attn);
    }
}

// ---------------------------------------------------------------------------
// Node update MLP: weights in LDS, grid-stride over 16-node tiles.
// ---------------------------------------------------------------------------
__global__ __launch_bounds__(128) void node_mlp_kernel(
    const float* __restrict__ x, const float* __restrict__ agg,
    const _Float16* __restrict__ w1f, const float* __restrict__ b1,
    const _Float16* __restrict__ w2f, const float* __restrict__ b2,
    float* __restrict__ out, int N)
{
    extern __shared__ char dsmem[];
    _Float16* ldsB1 = (_Float16*)dsmem;                                  // 32768 B
    _Float16* ldsB2 = (_Float16*)(dsmem + NODE_LDS_B1);                  // 16384 B
    _Float16* ldsIn = (_Float16*)(dsmem + NODE_LDS_B1 + NODE_LDS_B2);    // 16384 B
    _Float16* ldsU  = (_Float16*)(dsmem + NODE_LDS_B1 + NODE_LDS_B2 + NODE_LDS_IN);

    const int lane = threadIdx.x & 31;
    const int wid  = threadIdx.x >> 5;
    _Float16* inBase = ldsIn + (size_t)wid * 16 * 128;   // wave-private
    _Float16* uBase  = ldsU  + (size_t)wid * 16 * 128;   // wave-private

    // stage both weight fragment sets into LDS
    {
        const uint4* s1 = (const uint4*)w1f;
        uint4*       d1 = (uint4*)ldsB1;
        for (int i = threadIdx.x; i < NODE_LDS_B1 / 16; i += blockDim.x) d1[i] = s1[i];
        const uint4* s2 = (const uint4*)w2f;
        uint4*       d2 = (uint4*)ldsB2;
        for (int i = threadIdx.x; i < NODE_LDS_B2 / 16; i += blockDim.x) d2[i] = s2[i];
    }
    __syncthreads();

    const int ncol = lane & 15;
    const int kh   = (lane >> 4) << 3;
    const int m0   = (lane >> 4) << 3;

    float b1v[UPD_NT1], b2v[UPD_NT2];
    #pragma unroll
    for (int t = 0; t < UPD_NT1; ++t) b1v[t] = b1[t * 16 + ncol];
    #pragma unroll
    for (int t = 0; t < UPD_NT2; ++t) b2v[t] = b2[t * 16 + ncol];

    const long ntiles = ((long)N + 15) >> 4;
    for (long tile = (long)blockIdx.x * 4 + wid; tile < ntiles; tile += (long)gridDim.x * 4) {
        const long nb = tile << 4;

        // gather [x | agg] rows -> f16 LDS (128 halves per row)
        for (int m = 0; m < 16; ++m) {
            long node = nb + m;
            float4 p = make_float4(0.f, 0.f, 0.f, 0.f);
            if (node < N) {
                p = (lane < 16) ? *(const float4*)(x   + node * 64 + lane * 4)
                                : *(const float4*)(agg + node * 64 + (lane - 16) * 4);
            }
            *(v4h*)(inBase + m * 128 + lane * 4) = cvt4(p);
        }

        AF a[UPD_KC];
        const _Float16* arow = inBase + ncol * 128;
        #pragma unroll
        for (int k = 0; k < UPD_KC; ++k) {
            a[k].h[0] = *(const v8h*)(arow + k * 32 + kh);
            a[k].h[1] = *(const v8h*)(arow + k * 32 + kh + 16);
        }

        // layer 1: u = relu(in @ W1^T + b1) -> f16 LDS
        #pragma unroll
        for (int t = 0; t < UPD_NT1; ++t) {
            CF c; c.v = (v8f){};
            #pragma unroll
            for (int k = 0; k < UPD_KC; ++k) {
                const v16h* bp = (const v16h*)(ldsB1 + ((t * UPD_KC + k) * 512 + lane * 16));
                c.v = __builtin_amdgcn_wmma_f32_16x16x32_f16(
                          false, a[k].v, false, *bp, (short)0, c.v, false, false);
            }
            const int n = t * 16 + ncol;
            #pragma unroll
            for (int r = 0; r < 8; ++r) {
                float v = c.f[r] + b1v[t];
                uBase[(m0 + r) * 128 + n] = (_Float16)(v > 0.f ? v : 0.f);
            }
        }

        AF a2[UPD_KC];
        const _Float16* urow = uBase + ncol * 128;
        #pragma unroll
        for (int k = 0; k < UPD_KC; ++k) {
            a2[k].h[0] = *(const v8h*)(urow + k * 32 + kh);
            a2[k].h[1] = *(const v8h*)(urow + k * 32 + kh + 16);
        }

        // layer 2: out = relu(u @ W2^T + b2)
        #pragma unroll
        for (int t = 0; t < UPD_NT2; ++t) {
            CF c; c.v = (v8f){};
            #pragma unroll
            for (int k = 0; k < UPD_KC; ++k) {
                const v16h* bp = (const v16h*)(ldsB2 + ((t * UPD_KC + k) * 512 + lane * 16));
                c.v = __builtin_amdgcn_wmma_f32_16x16x32_f16(
                          false, a2[k].v, false, *bp, (short)0, c.v, false, false);
            }
            const int n = t * 16 + ncol;
            #pragma unroll
            for (int r = 0; r < 8; ++r) {
                long node = nb + m0 + r;
                if (node < N) {
                    float v = c.f[r] + b2v[t];
                    out[node * 64 + n] = v > 0.f ? v : 0.f;
                }
            }
        }
    }
}

extern "C" void kernel_launch(void* const* d_in, const int* in_sizes, int n_in,
                              void* d_out, int out_size, void* d_ws, size_t ws_size,
                              hipStream_t stream) {
    (void)n_in; (void)out_size; (void)ws_size;
    const float* x      = (const float*)d_in[0];
    const float* xs     = (const float*)d_in[1];
    const int*   ei     = (const int*)d_in[2];
    const float* efeat  = (const float*)d_in[3];
    const float* att_w1 = (const float*)d_in[4];
    const float* att_b1 = (const float*)d_in[5];
    const float* att_w2 = (const float*)d_in[6];
    const float* att_b2 = (const float*)d_in[7];
    const float* upd_w1 = (const float*)d_in[8];
    const float* upd_b1 = (const float*)d_in[9];
    const float* upd_w2 = (const float*)d_in[10];
    const float* upd_b2 = (const float*)d_in[11];
    float* out = (float*)d_out;

    const int N = in_sizes[0] / HDIM;   // 50000
    const int E = in_sizes[3] / 16;     // 800000

    // workspace carve-up
    char* ws = (char*)d_ws;
    size_t off = 0;
    _Float16* attw1f = (_Float16*)(ws + off); off += (size_t)ATT_NT * ATT_KC * 512 * 2;
    _Float16* updw1f = (_Float16*)(ws + off); off += (size_t)UPD_NT1 * UPD_KC * 512 * 2;
    _Float16* updw2f = (_Float16*)(ws + off); off += (size_t)UPD_NT2 * UPD_KC * 512 * 2;
    off = (off + 255) & ~(size_t)255;
    float* raww = (float*)(ws + off); off += (size_t)E * 4;
    float* nmax = (float*)(ws + off); off += (size_t)N * 4;
    float* nsum = (float*)(ws + off); off += (size_t)N * 4;
    float* agg  = (float*)(ws + off); off += (size_t)N * HDIM * 4;

    // 1) pack weights into f16 WMMA B-fragments
    {
        int tot1 = ATT_NT * ATT_KC * 512;
        pack_weights<<<(tot1 + 255) / 256, 256, 0, stream>>>(att_w1, ATT_W, ATT_IN, ATT_KC, tot1, attw1f);
        int tot2 = UPD_NT1 * UPD_KC * 512;
        pack_weights<<<(tot2 + 255) / 256, 256, 0, stream>>>(upd_w1, 128, 128, UPD_KC, tot2, updw1f);
        int tot3 = UPD_NT2 * UPD_KC * 512;
        pack_weights<<<(tot3 + 255) / 256, 256, 0, stream>>>(upd_w2, 64, 128, UPD_KC, tot3, updw2f);
    }

    // 2) init node accumulators
    {
        long tot = (long)N * HDIM;
        init_nodes<<<(unsigned)((tot + 255) / 256), 256, 0, stream>>>(nmax, nsum, agg, N);
    }

    // 3) edge attention MLP (WMMA, weights in LDS, grid-stride)
    {
        long tiles = ((long)E + 15) / 16;
        long want  = (tiles + 3) / 4;
        unsigned blocks = (unsigned)(want < 1024 ? want : 1024);
        edge_att_kernel<<<blocks, 128, EDGE_LDS_BYTES, stream>>>(
            x, xs, ei, efeat, attw1f, att_b1, att_w2, att_b2, raww, nmax, E);
    }

    // 4) exp + segment sum
    edge_exp_kernel<<<(unsigned)(((long)E + 255) / 256), 256, 0, stream>>>(ei, raww, nmax, nsum, E);

    // 5) normalized weighted scatter-sum
    edge_agg_kernel<<<(unsigned)(((long)E * 32 + 255) / 256), 256, 0, stream>>>(ei, raww, nsum, x, agg, E);

    // 6) node update MLP (WMMA, weights in LDS, grid-stride)
    {
        long tiles = ((long)N + 15) / 16;
        long want  = (tiles + 3) / 4;
        unsigned blocks = (unsigned)(want < 512 ? want : 512);
        node_mlp_kernel<<<blocks, 128, NODE_LDS_BYTES, stream>>>(
            x, agg, updw1f, upd_b1, updw2f, upd_b2, out, N);
    }
}
